// NRBS_30365418783271
// MI455X (gfx1250) — matrix operand order repaired and checked by reading, compiled, not run on hardware
//
#include <hip/hip_runtime.h>

#define NN   131072
#define NLAT 16
#define MUU  32
#define MGRP 32
#define BB   8
#define GG   (NN / MGRP)   // 4096 nodes per group

typedef __attribute__((ext_vector_type(16))) __bf16       bf16x16;
typedef __attribute__((ext_vector_type(8)))  __bf16       bf16x8;
typedef __attribute__((ext_vector_type(8)))  float        f32x8;
typedef __attribute__((ext_vector_type(4)))  int          i32x4;
typedef __attribute__((ext_vector_type(4)))  unsigned int u32x4;

// float -> bf16 (round to nearest even), bit-exact and toolchain-agnostic
__device__ __forceinline__ unsigned short f2bf(float f) {
    unsigned int u = __float_as_uint(f);
    unsigned int r = (u + 0x7FFFu + ((u >> 16) & 1u)) >> 16;
    return (unsigned short)r;
}

// ---------------------------------------------------------------------------
// K1: enc partials.  grid=(64 chunks, 8 batch), block=256.
// partial[(chunk*8+b)*16 + i] = sum over 2048 nodes of x[b,n]*W_enc[i,n]
// Deterministic LDS tree reduction (no float atomics).
// ---------------------------------------------------------------------------
__global__ void nrbs_encode(const float* __restrict__ x,
                            const float* __restrict__ Wenc,
                            float* __restrict__ partial) {
    const int b     = blockIdx.y;
    const int chunk = blockIdx.x;
    const int base  = chunk * 2048;
    const int t     = threadIdx.x;

    float acc[NLAT];
#pragma unroll
    for (int i = 0; i < NLAT; ++i) acc[i] = 0.f;

    for (int it = 0; it < 8; ++it) {
        const int node = base + it * 256 + t;
        const float xv = x[(size_t)b * NN + node];
#pragma unroll
        for (int i = 0; i < NLAT; ++i)
            acc[i] += xv * Wenc[(size_t)i * NN + node];
    }

    __shared__ float sh[256 * NLAT];
#pragma unroll
    for (int i = 0; i < NLAT; ++i) sh[i * 256 + t] = acc[i];
    __syncthreads();
    for (int off = 128; off > 0; off >>= 1) {
        if (t < off) {
#pragma unroll
            for (int i = 0; i < NLAT; ++i)
                sh[i * 256 + t] += sh[i * 256 + t + off];
        }
        __syncthreads();
    }
    if (t < NLAT)
        partial[(chunk * 8 + b) * NLAT + t] = sh[t * 256];
}

// ---------------------------------------------------------------------------
// K2: decoder [16, N] f32  ->  decT [N, 16] bf16  (32 B per node, contiguous)
// ---------------------------------------------------------------------------
__global__ void nrbs_dect(const float* __restrict__ dec,
                          unsigned short* __restrict__ decT) {
    const int node = blockIdx.x * 256 + threadIdx.x;
    unsigned short v[16];
#pragma unroll
    for (int i = 0; i < NLAT; ++i) v[i] = f2bf(dec[(size_t)i * NN + node]);
    u32x4 p0, p1;
    p0.x = (unsigned)v[0]  | ((unsigned)v[1]  << 16);
    p0.y = (unsigned)v[2]  | ((unsigned)v[3]  << 16);
    p0.z = (unsigned)v[4]  | ((unsigned)v[5]  << 16);
    p0.w = (unsigned)v[6]  | ((unsigned)v[7]  << 16);
    p1.x = (unsigned)v[8]  | ((unsigned)v[9]  << 16);
    p1.y = (unsigned)v[10] | ((unsigned)v[11] << 16);
    p1.z = (unsigned)v[12] | ((unsigned)v[13] << 16);
    p1.w = (unsigned)v[14] | ((unsigned)v[15] << 16);
    u32x4* dst = (u32x4*)(decT + (size_t)node * 16);
    dst[0] = p0;
    dst[1] = p1;
}

// ---------------------------------------------------------------------------
// K3: build WMMA B-operand table.
// Bbuf[m][k][b16][i16] (bf16) = enc[b,i] * win[b,i,m,k], zero for b>=8.
// grid = 32 (one per group m), block = 256.
// ---------------------------------------------------------------------------
__global__ void nrbs_bbuf(const float* __restrict__ partial,
                          const float* __restrict__ b_enc,
                          const float* __restrict__ Wbw,
                          const float* __restrict__ bbw,
                          unsigned short* __restrict__ Bbuf) {
    const int m = blockIdx.x;
    const int t = threadIdx.x;
    __shared__ float encS[BB * NLAT];

    if (t < BB * NLAT) {
        const int b = t >> 4, i = t & 15;
        float s = b_enc[i];
        for (int c = 0; c < 64; ++c)          // fixed order -> deterministic
            s += partial[(c * 8 + b) * NLAT + i];
        encS[t] = s;
    }
    __syncthreads();

    const int i = t & 15;
    if (t < BB * NLAT) {
        const int b = t >> 4;
        float z = bbw[i * MGRP + m];
#pragma unroll
        for (int j = 0; j < NLAT; ++j)
            z += encS[b * NLAT + j] * Wbw[((size_t)i * MGRP + m) * NLAT + j];
        const float bw   = 1.f / (1.f + __expf(-z));
        const float w    = bw * (float)MUU;
        const float inv2 = 1.f / (w * w);
        float win[MUU];
        float sum = 0.f;
#pragma unroll
        for (int k = 0; k < MUU; ++k) {
            float v = 1.f - (float)(k * k) * inv2;
            v = v > 0.f ? v : 0.f;
            win[k] = v;
            sum += v;
        }
        const float scale = encS[b * NLAT + i] / sum;
#pragma unroll
        for (int k = 0; k < MUU; ++k)
            Bbuf[(((size_t)m * MUU + k) * 16 + b) * 16 + i] = f2bf(win[k] * scale);
    } else {
        const int b2 = 8 + ((t - 128) >> 4);
#pragma unroll
        for (int k = 0; k < MUU; ++k)
            Bbuf[(((size_t)m * MUU + k) * 16 + b2) * 16 + i] = 0;
    }
}

// ---------------------------------------------------------------------------
// K4: main fused gather + WMMA GEMM + decode.
// grid = 1024 blocks x 256 threads (8 waves); wave w owns the 16-node tile
// starting at blk*128 + w*16; all tiles of a block share group m = blk/32.
// Out[node, b] = sum_s  A_s[16x32] x B_s[32x16], s = 0..15 (K = 512 total).
//   A_s[M=node, K] = decT[nb[node, 2s + K/16]][K%16]   (bf16 gather, b128 loads)
//   B_s[K, N=b]    = Bbuf[m][2s + K/16][b][K%16]        (from LDS)
// All per-lane addresses are 32-bit offsets from uniform bases so the
// compiler emits SADDR-form global loads/stores (GVS addressing).
// ---------------------------------------------------------------------------
__global__ void __launch_bounds__(256) nrbs_smooth(
    const unsigned short* __restrict__ decT,
    const int*            __restrict__ nbrs,
    const unsigned short* __restrict__ Bbuf,
    float*                __restrict__ out) {
    __shared__ __align__(32) unsigned short ldsB[MUU * 16 * 16];  // 16 KB

    const int blk = blockIdx.x;
    const int m   = blk >> 5;

    // stage this group's B table into LDS (16 KB, 4 x b128 per thread)
    {
        const char* srcB = (const char*)Bbuf + (unsigned)m * (MUU * 16 * 16 * 2u);
        const unsigned so = (unsigned)threadIdx.x * 16u;
        u32x4* dst = (u32x4*)ldsB;
#pragma unroll
        for (int q = 0; q < 4; ++q)
            dst[threadIdx.x + 256 * q] = *(const u32x4*)(srcB + so + (unsigned)q * 4096u);
    }
    __syncthreads();

    const int lane     = threadIdx.x & 31;
    const int wave     = threadIdx.x >> 5;
    const int half     = lane >> 4;     // 0: K-low half, 1: K-high half
    const int row      = lane & 15;     // A row (node) and B column (batch)
    const int tileBase = blk * 128 + wave * 16;
    const int node     = tileBase + row;

    // preload the 32 neighbour indices of this lane's node (32-bit offset base)
    const char* nbrB = (const char*)nbrs;
    const unsigned nOff = (unsigned)node << 7;   // node * 32 * 4 bytes
    i32x4 nb[8];
#pragma unroll
    for (int q = 0; q < 8; ++q)
        nb[q] = *(const i32x4*)(nbrB + nOff + (unsigned)q * 16u);

    const char*   decTb   = (const char*)decT;
    const unsigned halfOff = (unsigned)half * 16u;   // byte offset within node row

    f32x8 c = {0.f, 0.f, 0.f, 0.f, 0.f, 0.f, 0.f, 0.f};

#pragma unroll
    for (int s = 0; s < 16; ++s) {
        const unsigned idx0 = (unsigned)nb[s >> 1][(s & 1) * 2 + half];
        const unsigned idx1 = idx0;
        const unsigned off0 = (idx0 << 5);               // idx * 32 B (latents 0..7)
        const unsigned off1 = (idx1 << 5) + 16u;         // latents 8..15
        // A fragment: two contiguous 16-byte bf16 gathers per lane (saddr form)
        bf16x8 a0 = *(const bf16x8*)(decTb + off0);
        bf16x8 a1 = *(const bf16x8*)(decTb + off1);
        bf16x16 a;
#pragma unroll
        for (int q = 0; q < 8; ++q) { a[q] = a0[q]; a[8 + q] = a1[q]; }
        // B fragment: one contiguous 32-byte LDS read
        const int kRow = 2 * s + half;
        bf16x16 bm = *(const bf16x16*)(ldsB + ((size_t)kRow * 16 + row) * 16);

        c = __builtin_amdgcn_wmma_f32_16x16x32_bf16(
                /*neg_a=*/false, a, /*neg_b=*/false, bm,
                /*c_mod=*/(short)0, c, /*reuse_a=*/false, /*reuse_b=*/false);
    }

    // D layout: VGPR d -> M = d + 8*half (node row), N = lane%16 (batch col)
    if (row < BB) {
        char* outB = (char*)out;
        const unsigned bBase = ((unsigned)row * NN + (unsigned)tileBase +
                                (unsigned)half * 8u) * 4u;
#pragma unroll
        for (int d = 0; d < 8; ++d)
            *(float*)(outB + bBase + (unsigned)d * 4u) = c[d];
    }
}

// ---------------------------------------------------------------------------
extern "C" void kernel_launch(void* const* d_in, const int* in_sizes, int n_in,
                              void* d_out, int out_size, void* d_ws, size_t ws_size,
                              hipStream_t stream) {
    const float* x       = (const float*)d_in[0];   // [B, N]
    const float* Wenc    = (const float*)d_in[1];   // [n, N]
    const float* b_enc   = (const float*)d_in[2];   // [n]
    const float* Wbw     = (const float*)d_in[3];   // [n, M, n]
    const float* bbw     = (const float*)d_in[4];   // [n, M]
    const float* decoder = (const float*)d_in[5];   // [n, N]
    const int*   nbrs    = (const int*)d_in[6];     // [N, MU]
    float*       out     = (float*)d_out;           // [B, N]

    char* ws = (char*)d_ws;
    float*          partial = (float*)ws;                                   // 64*8*16*4 = 32 KB
    unsigned short* decT    = (unsigned short*)(ws + 65536);                // N*16*2    = 4 MB
    unsigned short* Bbuf    = (unsigned short*)(ws + 65536 + (size_t)NN * 16 * 2); // 512 KB

    nrbs_encode<<<dim3(64, 8), 256, 0, stream>>>(x, Wenc, partial);
    nrbs_dect  <<<NN / 256,    256, 0, stream>>>(decoder, decT);
    nrbs_bbuf  <<<MGRP,        256, 0, stream>>>(partial, b_enc, Wbw, bbw, Bbuf);
    nrbs_smooth<<<NN / 128,    256, 0, stream>>>(decT, nbrs, Bbuf, out);
}
